// MPNN_17686675325408
// MI455X (gfx1250) — matrix operand order repaired and checked by reading, not compiled, not run on hardware
//
#include <hip/hip_runtime.h>
#include <math.h>

typedef __attribute__((ext_vector_type(16))) _Float16 v16h;
typedef __attribute__((ext_vector_type(8)))  float    v8f;

#define NNODES 50000
#define NEDGES 800000
#define INDIM  256
#define HDIM   128
#define NLAYERS 4
#define NCLS   230
#define NCLSP  240
#define KRBF   16
#define WYSTART 128
#define WYDIM   26
#define NBINS  4096

// ---------------- device helpers ----------------

__device__ __forceinline__ v8f vzero8() {
  v8f z = {0.f,0.f,0.f,0.f,0.f,0.f,0.f,0.f};
  return z;
}

__device__ __forceinline__ float siluf(float x) {
  return x / (1.0f + __expf(-x));
}

__device__ __forceinline__ v8f wmma_f16(v16h a, v16h b, v8f c) {
  return __builtin_amdgcn_wmma_f32_16x16x32_f16(false, a, false, b, (short)0, c, false, false);
}

// A-fragment 16x32 f16 (ISA layout): lane = m + 16*g.
// pairBase = k0/2 + 4*g (in 2-half units). row must be 4B aligned.
__device__ __forceinline__ v16h loadA_h(const _Float16* row, int pairBase) {
  const unsigned int* p = (const unsigned int*)row;
  union { v16h v; unsigned int u[8]; } r;
  r.u[0] = p[pairBase + 0];  r.u[1] = p[pairBase + 1];
  r.u[2] = p[pairBase + 2];  r.u[3] = p[pairBase + 3];
  r.u[4] = p[pairBase + 8];  r.u[5] = p[pairBase + 9];
  r.u[6] = p[pairBase + 10]; r.u[7] = p[pairBase + 11];
  return r.v;
}

// Same A fragment but source row is f32 (convert on load).
__device__ __forceinline__ v16h loadA_f(const float* row, int k0, int g) {
  union { v16h v; _Float16 h[16]; } r;
  int b = k0 + 8 * g;
#pragma unroll
  for (int j = 0; j < 4; ++j) {
    r.h[2*j]   = (_Float16)row[b + 2*j];
    r.h[2*j+1] = (_Float16)row[b + 2*j + 1];
  }
#pragma unroll
  for (int j = 0; j < 4; ++j) {
    r.h[8+2*j] = (_Float16)row[b + 16 + 2*j];
    r.h[9+2*j] = (_Float16)row[b + 16 + 2*j + 1];
  }
  return r.v;
}

// B fragments pre-packed fragment-contiguous: one v16h (32B) per lane.
__device__ __forceinline__ v16h loadB(const _Float16* packed, int frag, int lane) {
  const v16h* p = (const v16h*)packed;
  return p[(size_t)frag * 32 + lane];
}

__device__ __forceinline__ void atomicAddF32(float* p, float v) {
#if defined(__gfx1250__)
  asm volatile("global_atomic_add_f32 %0, %1, off" :: "v"(p), "v"(v) : "memory");
#else
  atomicAdd(p, v);
#endif
}

// ---------------- weight packing ----------------
// Packs f32 row-major W[K x ld] (valid region Kvalid x Nvalid) into B-fragment
// layout: frag f = t*ntiles + nt; per lane 16 halves; B mapping k = t*32 + 8g + 2j + (j >= 4 ? 8 : 0).
__global__ void pack_b_kernel(const float* __restrict__ W, int ld, int Kvalid,
                              int Nvalid, int ntiles, _Float16* __restrict__ out) {
  int frag = blockIdx.x;
  int lane = threadIdx.x;
  int t = frag / ntiles, nt = frag % ntiles;
  int g = lane >> 4, ln = lane & 15;
  int n = nt * 16 + ln;
  _Float16* o = out + ((size_t)frag * 32 + lane) * 16;
#pragma unroll
  for (int j = 0; j < 8; ++j) {
    int k = t * 32 + 8 * g + 2 * j + (j >= 4 ? 8 : 0);
    float v0 = (k < Kvalid && n < Nvalid) ? W[(size_t)k * ld + n] : 0.0f;
    float v1 = (k + 1 < Kvalid && n < Nvalid) ? W[(size_t)(k + 1) * ld + n] : 0.0f;
    o[2*j]   = (_Float16)v0;
    o[2*j+1] = (_Float16)v1;
  }
}

// ---------------- small utility kernels ----------------

__global__ void zero_kernel(float* __restrict__ p, long long n) {
  long long i = (long long)blockIdx.x * 256 + threadIdx.x;
  if (i < n) p[i] = 0.0f;
}

__global__ void gid_kernel(const float* __restrict__ x, int* __restrict__ gid,
                           int* __restrict__ gcnt) {
  int n = blockIdx.x * 256 + threadIdx.x;
  if (n >= NNODES) return;
  const float* row = x + (size_t)n * INDIM + WYSTART;
  int best = 0; float bv = row[0];
#pragma unroll
  for (int k = 1; k < WYDIM; ++k) {
    float v = row[k];
    if (v > bv) { bv = v; best = k; }
  }
  gid[n] = best;
  atomicAdd(&gcnt[best], 1);
}

__global__ void hist_kernel(const float* __restrict__ ea, int* __restrict__ hist) {
  int e = blockIdx.x * 256 + threadIdx.x;
  if (e >= NEDGES) return;
  float vx = ea[(size_t)e*3+0], vy = ea[(size_t)e*3+1], vz = ea[(size_t)e*3+2];
  float r = fmaxf(sqrtf(vx*vx + vy*vy + vz*vz), 1e-8f);
  int b = (int)(r * (NBINS / 8.0f));
  b = b < 0 ? 0 : (b > NBINS - 1 ? NBINS - 1 : b);
  atomicAdd(&hist[b], 1);
}

__global__ void rmax_kernel(const int* __restrict__ hist, float* __restrict__ rmax) {
  if (threadIdx.x != 0) return;
  long long target = (long long)(0.95 * (double)(NEDGES - 1));
  long long cum = 0; int bi = NBINS - 1;
  for (int i = 0; i < NBINS; ++i) {
    cum += hist[i];
    if (cum > target) { bi = i; break; }
  }
  float r = (bi + 0.5f) * (8.0f / NBINS);
  rmax[0] = fminf(fmaxf(r, 1.0f), 8.0f);
}

__global__ void out_kernel(const int* __restrict__ gid, const float* __restrict__ gsum,
                           const int* __restrict__ gcnt, float* __restrict__ out) {
  size_t i = (size_t)blockIdx.x * 256 + threadIdx.x;
  if (i >= (size_t)NNODES * NCLS) return;
  int n = (int)(i / NCLS), c = (int)(i % NCLS);
  int g = gid[n];
  float cnt = fmaxf((float)gcnt[g], 1.0f);
  out[i] = gsum[(size_t)g * NCLSP + c] / cnt;
}

// ---------------- node encoder: h0 = silu(LN(x@W_in + b)) ----------------

__global__ __launch_bounds__(256) void encoder_kernel(
    const float* __restrict__ x, const _Float16* __restrict__ Wp,
    const float* __restrict__ bin, const float* __restrict__ gin,
    const float* __restrict__ bein, float* __restrict__ h32,
    _Float16* __restrict__ state0) {
  int lane = threadIdx.x & 31, wave = threadIdx.x >> 5;
  int tile = blockIdx.x * 8 + wave;
  if (tile >= NNODES / 16) return;
  int g = lane >> 4, ln = lane & 15;
  const float* arow = x + (size_t)(tile * 16 + ln) * INDIM;

  v8f c[8];
#pragma unroll
  for (int i = 0; i < 8; ++i) c[i] = vzero8();
#pragma unroll
  for (int t = 0; t < 8; ++t) {
    v16h a = loadA_f(arow, t * 32, g);
#pragma unroll
    for (int nt = 0; nt < 8; ++nt)
      c[nt] = wmma_f16(a, loadB(Wp, t * 8 + nt, lane), c[nt]);
  }
  float vals[8][8];
#pragma unroll
  for (int nt = 0; nt < 8; ++nt) {
    float bb = bin[nt * 16 + ln];
#pragma unroll
    for (int r = 0; r < 8; ++r) vals[r][nt] = c[nt][r] + bb;
  }
#pragma unroll
  for (int r = 0; r < 8; ++r) {
    float s = 0.f, s2 = 0.f;
#pragma unroll
    for (int nt = 0; nt < 8; ++nt) { float v = vals[r][nt]; s += v; s2 += v * v; }
#pragma unroll
    for (int msk = 1; msk < 16; msk <<= 1) {
      s  += __shfl_xor(s, msk, 32);
      s2 += __shfl_xor(s2, msk, 32);
    }
    float mu = s * (1.0f / HDIM);
    float rinv = rsqrtf(s2 * (1.0f / HDIM) - mu * mu + 1e-5f);
    size_t nrow = (size_t)(tile * 16 + r + 8 * g);
#pragma unroll
    for (int nt = 0; nt < 8; ++nt) {
      int col = nt * 16 + ln;
      float v = siluf((vals[r][nt] - mu) * rinv * gin[col] + bein[col]);
      h32[nrow * HDIM + col] = v;
      state0[nrow * HDIM + col] = (_Float16)v;
    }
  }
}

// ---------------- edge encoder: e_enc = LN(silu(e @ W_e + b_e)) ----------------

__global__ __launch_bounds__(256) void edge_enc_kernel(
    const float* __restrict__ ea, const float* __restrict__ rmaxp,
    const _Float16* __restrict__ Wp, const float* __restrict__ be,
    const float* __restrict__ ge, const float* __restrict__ bee,
    _Float16* __restrict__ e16) {
  __shared__ _Float16 feat[8][16][32];
  int lane = threadIdx.x & 31, wave = threadIdx.x >> 5;
  int tile = blockIdx.x * 8 + wave;
  if (tile >= NEDGES / 16) return;
  int g = lane >> 4, ln = lane & 15;
  float r_max = rmaxp[0];

  if (lane < 16) {
    int e = tile * 16 + lane;
    float vx = ea[(size_t)e*3+0], vy = ea[(size_t)e*3+1], vz = ea[(size_t)e*3+2];
    float r = fmaxf(sqrtf(vx*vx + vy*vy + vz*vz), 1e-8f);
    float inv = 1.0f / r;
    _Float16* row = feat[wave][lane];
    row[0] = (_Float16)(vx * inv);
    row[1] = (_Float16)(vy * inv);
    row[2] = (_Float16)(vz * inv);
    row[3] = (_Float16)r;
    float delta = fmaxf(r_max / (KRBF - 1), 0.001f);
    float gamma = 1.0f / (2.0f * (0.5f * delta) * (0.5f * delta));
#pragma unroll
    for (int k = 0; k < KRBF; ++k) {
      float ck = r_max * ((float)k / (KRBF - 1));
      float d = r - ck;
      row[4 + k] = (_Float16)__expf(-gamma * d * d);
    }
#pragma unroll
    for (int k = 20; k < 32; ++k) row[k] = (_Float16)0.0f;
  }

  v8f c[8];
#pragma unroll
  for (int i = 0; i < 8; ++i) c[i] = vzero8();
  v16h a = loadA_h(&feat[wave][ln][0], 4 * g);
#pragma unroll
  for (int nt = 0; nt < 8; ++nt)
    c[nt] = wmma_f16(a, loadB(Wp, nt, lane), c[nt]);

  float vals[8][8];
#pragma unroll
  for (int nt = 0; nt < 8; ++nt) {
    float bb = be[nt * 16 + ln];
#pragma unroll
    for (int r = 0; r < 8; ++r) vals[r][nt] = siluf(c[nt][r] + bb);
  }
#pragma unroll
  for (int r = 0; r < 8; ++r) {
    float s = 0.f, s2 = 0.f;
#pragma unroll
    for (int nt = 0; nt < 8; ++nt) { float v = vals[r][nt]; s += v; s2 += v * v; }
#pragma unroll
    for (int msk = 1; msk < 16; msk <<= 1) {
      s  += __shfl_xor(s, msk, 32);
      s2 += __shfl_xor(s2, msk, 32);
    }
    float mu = s * (1.0f / HDIM);
    float rinv = rsqrtf(s2 * (1.0f / HDIM) - mu * mu + 1e-5f);
    size_t erow = (size_t)(tile * 16 + r + 8 * g);
#pragma unroll
    for (int nt = 0; nt < 8; ++nt) {
      int col = nt * 16 + ln;
      e16[erow * HDIM + col] =
          (_Float16)((vals[r][nt] - mu) * rinv * ge[col] + bee[col]);
    }
  }
}

// -------- per-layer message MLP + scatter-add segment sum --------
// m1 = silu([h[src]|h[dst]|e_enc] @ W1 + b1); m2 = silu(m1 @ W2 + b2); m[dst] += m2

__global__ __launch_bounds__(256) void msg_kernel(
    const int* __restrict__ src, const int* __restrict__ dst,
    const _Float16* __restrict__ hcur, const _Float16* __restrict__ e16,
    const _Float16* __restrict__ W1p, const float* __restrict__ b1,
    const _Float16* __restrict__ W2p, const float* __restrict__ b2,
    float* __restrict__ m) {
  __shared__ _Float16 mid[8][16][HDIM];
  int lane = threadIdx.x & 31, wave = threadIdx.x >> 5;
  int tile = blockIdx.x * 8 + wave;
  if (tile >= NEDGES / 16) return;
  int g = lane >> 4, ln = lane & 15;
  int erow = tile * 16 + ln;
  const _Float16* hs = hcur + (size_t)src[erow] * HDIM;
  const _Float16* hd = hcur + (size_t)dst[erow] * HDIM;
  const _Float16* ee = e16 + (size_t)erow * HDIM;

  v8f c[8];
#pragma unroll
  for (int i = 0; i < 8; ++i) c[i] = vzero8();
#pragma unroll
  for (int t = 0; t < 12; ++t) {
    const _Float16* arow = (t < 4) ? hs : (t < 8) ? hd : ee;
    int pairBase = (t & 3) * 16 + 4 * g;
    v16h a = loadA_h(arow, pairBase);
#pragma unroll
    for (int nt = 0; nt < 8; ++nt)
      c[nt] = wmma_f16(a, loadB(W1p, t * 8 + nt, lane), c[nt]);
  }
  // silu + stage into LDS (C-layout -> row-major f16)
#pragma unroll
  for (int nt = 0; nt < 8; ++nt) {
    float bb = b1[nt * 16 + ln];
#pragma unroll
    for (int r = 0; r < 8; ++r)
      mid[wave][r + 8 * g][nt * 16 + ln] = (_Float16)siluf(c[nt][r] + bb);
  }
  // second GEMM, K=128
  v8f d2[8];
#pragma unroll
  for (int i = 0; i < 8; ++i) d2[i] = vzero8();
  const _Float16* arow2 = &mid[wave][ln][0];
#pragma unroll
  for (int t = 0; t < 4; ++t) {
    v16h a = loadA_h(arow2, t * 16 + 4 * g);
#pragma unroll
    for (int nt = 0; nt < 8; ++nt)
      d2[nt] = wmma_f16(a, loadB(W2p, t * 8 + nt, lane), d2[nt]);
  }
  // silu + scatter-add to m[dst]
  float bb2[8];
#pragma unroll
  for (int nt = 0; nt < 8; ++nt) bb2[nt] = b2[nt * 16 + ln];
#pragma unroll
  for (int r = 0; r < 8; ++r) {
    int dd = dst[tile * 16 + r + 8 * g];
    float* mrow = m + (size_t)dd * HDIM;
#pragma unroll
    for (int nt = 0; nt < 8; ++nt)
      atomicAddF32(&mrow[nt * 16 + ln], siluf(d2[nt][r] + bb2[nt]));
  }
}

// -------- per-layer update: h = LN(silu([h|m]@W + b) + h) --------

__global__ __launch_bounds__(256) void update_kernel(
    const _Float16* __restrict__ hcur, const float* __restrict__ m,
    float* __restrict__ h32, const _Float16* __restrict__ Wp,
    const float* __restrict__ ub, const float* __restrict__ ng,
    const float* __restrict__ nb, _Float16* __restrict__ stateOut) {
  int lane = threadIdx.x & 31, wave = threadIdx.x >> 5;
  int tile = blockIdx.x * 8 + wave;
  if (tile >= NNODES / 16) return;
  int g = lane >> 4, ln = lane & 15;
  const _Float16* hrow = hcur + (size_t)(tile * 16 + ln) * HDIM;
  const float* mrow = m + (size_t)(tile * 16 + ln) * HDIM;

  v8f c[8];
#pragma unroll
  for (int i = 0; i < 8; ++i) c[i] = vzero8();
#pragma unroll
  for (int t = 0; t < 8; ++t) {
    v16h a = (t < 4) ? loadA_h(hrow, t * 16 + 4 * g)
                     : loadA_f(mrow, (t - 4) * 32, g);
#pragma unroll
    for (int nt = 0; nt < 8; ++nt)
      c[nt] = wmma_f16(a, loadB(Wp, t * 8 + nt, lane), c[nt]);
  }
  float vals[8][8];
#pragma unroll
  for (int nt = 0; nt < 8; ++nt) {
    float bb = ub[nt * 16 + ln];
#pragma unroll
    for (int r = 0; r < 8; ++r) {
      size_t nrow = (size_t)(tile * 16 + r + 8 * g);
      float hold = h32[nrow * HDIM + nt * 16 + ln];
      vals[r][nt] = siluf(c[nt][r] + bb) + hold;
    }
  }
#pragma unroll
  for (int r = 0; r < 8; ++r) {
    float s = 0.f, s2 = 0.f;
#pragma unroll
    for (int nt = 0; nt < 8; ++nt) { float v = vals[r][nt]; s += v; s2 += v * v; }
#pragma unroll
    for (int msk = 1; msk < 16; msk <<= 1) {
      s  += __shfl_xor(s, msk, 32);
      s2 += __shfl_xor(s2, msk, 32);
    }
    float mu = s * (1.0f / HDIM);
    float rinv = rsqrtf(s2 * (1.0f / HDIM) - mu * mu + 1e-5f);
    size_t nrow = (size_t)(tile * 16 + r + 8 * g);
#pragma unroll
    for (int nt = 0; nt < 8; ++nt) {
      int col = nt * 16 + ln;
      float v = (vals[r][nt] - mu) * rinv * ng[col] + nb[col];
      h32[nrow * HDIM + col] = v;
      stateOut[nrow * HDIM + col] = (_Float16)v;
    }
  }
}

// -------- JK + output head, fused, with group atomic sums --------

__global__ __launch_bounds__(256) void head_kernel(
    const _Float16* __restrict__ states, const _Float16* __restrict__ Wjkp,
    const float* __restrict__ bjk, const _Float16* __restrict__ Wo1p,
    const float* __restrict__ bo1, const _Float16* __restrict__ Wo2p,
    const float* __restrict__ bo2, const int* __restrict__ gid,
    float* __restrict__ gsum) {
  __shared__ _Float16 buf[8][16][HDIM];
  int lane = threadIdx.x & 31, wave = threadIdx.x >> 5;
  int tile = blockIdx.x * 8 + wave;
  if (tile >= NNODES / 16) return;
  int g = lane >> 4, ln = lane & 15;
  int nodeA = tile * 16 + ln;

  // GEMM1: concat(states) @ W_jk  (K = 640)
  v8f c[8];
#pragma unroll
  for (int i = 0; i < 8; ++i) c[i] = vzero8();
#pragma unroll
  for (int t = 0; t < 20; ++t) {
    int l = t >> 2;
    const _Float16* arow = states + ((size_t)l * NNODES + nodeA) * HDIM;
    v16h a = loadA_h(arow, (t & 3) * 16 + 4 * g);
#pragma unroll
    for (int nt = 0; nt < 8; ++nt)
      c[nt] = wmma_f16(a, loadB(Wjkp, t * 8 + nt, lane), c[nt]);
  }
#pragma unroll
  for (int nt = 0; nt < 8; ++nt) {
    float bb = bjk[nt * 16 + ln];
#pragma unroll
    for (int r = 0; r < 8; ++r)
      buf[wave][r + 8 * g][nt * 16 + ln] = (_Float16)(c[nt][r] + bb);
  }
  // GEMM2: h_agg @ W_o1 -> silu (K = 128)
  v8f o[8];
#pragma unroll
  for (int i = 0; i < 8; ++i) o[i] = vzero8();
  const _Float16* arow2 = &buf[wave][ln][0];
#pragma unroll
  for (int t = 0; t < 4; ++t) {
    v16h a = loadA_h(arow2, t * 16 + 4 * g);
#pragma unroll
    for (int nt = 0; nt < 8; ++nt)
      o[nt] = wmma_f16(a, loadB(Wo1p, t * 8 + nt, lane), o[nt]);
  }
#pragma unroll
  for (int nt = 0; nt < 8; ++nt) {
    float bb = bo1[nt * 16 + ln];
#pragma unroll
    for (int r = 0; r < 8; ++r)
      buf[wave][r + 8 * g][nt * 16 + ln] = (_Float16)siluf(o[nt][r] + bb);
  }
  // GEMM3: @ W_o2 (N padded to 240, 15 tiles), then group scatter
  v8f z[15];
#pragma unroll
  for (int i = 0; i < 15; ++i) z[i] = vzero8();
#pragma unroll
  for (int t = 0; t < 4; ++t) {
    v16h a = loadA_h(arow2, t * 16 + 4 * g);
#pragma unroll
    for (int nt = 0; nt < 15; ++nt)
      z[nt] = wmma_f16(a, loadB(Wo2p, t * 15 + nt, lane), z[nt]);
  }
  int gids[8];
#pragma unroll
  for (int r = 0; r < 8; ++r) gids[r] = gid[tile * 16 + r + 8 * g];
#pragma unroll
  for (int nt = 0; nt < 15; ++nt) {
    int col = nt * 16 + ln;
    if (col < NCLS) {
      float bb = bo2[col];
#pragma unroll
      for (int r = 0; r < 8; ++r)
        atomicAddF32(&gsum[(size_t)gids[r] * NCLSP + col], z[nt][r] + bb);
    }
  }
}

// ---------------- host orchestration ----------------

extern "C" void kernel_launch(void* const* d_in, const int* in_sizes, int n_in,
                              void* d_out, int out_size, void* d_ws, size_t ws_size,
                              hipStream_t stream) {
  const float* x        = (const float*)d_in[0];
  const int*   eidx     = (const int*)d_in[1];
  const float* eattr    = (const float*)d_in[2];
  const float* W_in     = (const float*)d_in[3];
  const float* b_in     = (const float*)d_in[4];
  const float* g_in     = (const float*)d_in[5];
  const float* be_in    = (const float*)d_in[6];
  const float* W_e      = (const float*)d_in[7];
  const float* b_e      = (const float*)d_in[8];
  const float* g_e      = (const float*)d_in[9];
  const float* be_e     = (const float*)d_in[10];
  const float* msg_W1   = (const float*)d_in[11];
  const float* msg_b1   = (const float*)d_in[12];
  const float* msg_W2   = (const float*)d_in[13];
  const float* msg_b2   = (const float*)d_in[14];
  const float* upd_W    = (const float*)d_in[15];
  const float* upd_b    = (const float*)d_in[16];
  const float* norm_g   = (const float*)d_in[17];
  const float* norm_b   = (const float*)d_in[18];
  const float* W_jk     = (const float*)d_in[19];
  const float* b_jk     = (const float*)d_in[20];
  const float* W_o1     = (const float*)d_in[21];
  const float* b_o1     = (const float*)d_in[22];
  const float* W_o2     = (const float*)d_in[23];
  const float* b_o2     = (const float*)d_in[24];
  float* out = (float*)d_out;

  char* ws = (char*)d_ws;
  size_t off = 0;
  auto alloc = [&](size_t bytes) -> char* {
    char* p = ws + off;
    off = (off + bytes + 255) & ~(size_t)255;
    return p;
  };
  _Float16* e16    = (_Float16*)alloc((size_t)NEDGES * HDIM * 2);
  float*    m      = (float*)alloc((size_t)NNODES * HDIM * 4);
  _Float16* states = (_Float16*)alloc((size_t)(NLAYERS + 1) * NNODES * HDIM * 2);
  float*    h32    = (float*)alloc((size_t)NNODES * HDIM * 4);
  int*      gid    = (int*)alloc((size_t)NNODES * 4);
  int*      hist   = (int*)alloc((size_t)NBINS * 4);
  float*    rmax   = (float*)alloc(256);
  float*    gsum   = (float*)alloc((size_t)WYDIM * NCLSP * 4);
  int*      gcnt   = (int*)alloc((size_t)WYDIM * 4);
  _Float16* Wp_in = (_Float16*)alloc((size_t)8 * 8 * 512 * 2);
  _Float16* Wp_e  = (_Float16*)alloc((size_t)1 * 8 * 512 * 2);
  _Float16* Wp_m1 = (_Float16*)alloc((size_t)4 * 12 * 8 * 512 * 2);
  _Float16* Wp_m2 = (_Float16*)alloc((size_t)4 * 4 * 8 * 512 * 2);
  _Float16* Wp_u  = (_Float16*)alloc((size_t)4 * 8 * 8 * 512 * 2);
  _Float16* Wp_jk = (_Float16*)alloc((size_t)20 * 8 * 512 * 2);
  _Float16* Wp_o1 = (_Float16*)alloc((size_t)4 * 8 * 512 * 2);
  _Float16* Wp_o2 = (_Float16*)alloc((size_t)4 * 15 * 512 * 2);

  const int* src = eidx;
  const int* dst = eidx + NEDGES;

  // Pack all weight matrices into B-fragment layout (f16)
  pack_b_kernel<<<8 * 8, 32, 0, stream>>>(W_in, 128, 256, 128, 8, Wp_in);
  pack_b_kernel<<<1 * 8, 32, 0, stream>>>(W_e, 128, 20, 128, 8, Wp_e);
  for (int l = 0; l < NLAYERS; ++l) {
    pack_b_kernel<<<12 * 8, 32, 0, stream>>>(msg_W1 + (size_t)l * 384 * 128, 128,
                                             384, 128, 8, Wp_m1 + (size_t)l * 12 * 8 * 512);
    pack_b_kernel<<<4 * 8, 32, 0, stream>>>(msg_W2 + (size_t)l * 128 * 128, 128,
                                            128, 128, 8, Wp_m2 + (size_t)l * 4 * 8 * 512);
    pack_b_kernel<<<8 * 8, 32, 0, stream>>>(upd_W + (size_t)l * 256 * 128, 128,
                                            256, 128, 8, Wp_u + (size_t)l * 8 * 8 * 512);
  }
  pack_b_kernel<<<20 * 8, 32, 0, stream>>>(W_jk, 128, 640, 128, 8, Wp_jk);
  pack_b_kernel<<<4 * 8, 32, 0, stream>>>(W_o1, 128, 128, 128, 8, Wp_o1);
  pack_b_kernel<<<4 * 15, 32, 0, stream>>>(W_o2, 230, 128, 230, 15, Wp_o2);

  // Zero small accumulators
  zero_kernel<<<(NBINS + 255) / 256, 256, 0, stream>>>((float*)hist, NBINS);
  zero_kernel<<<(WYDIM * NCLSP + 255) / 256, 256, 0, stream>>>(gsum, WYDIM * NCLSP);
  zero_kernel<<<1, 256, 0, stream>>>((float*)gcnt, WYDIM);

  // Group ids + counts; r histogram; r_max (approx 0.95 quantile, clipped [1,8])
  gid_kernel<<<(NNODES + 255) / 256, 256, 0, stream>>>(x, gid, gcnt);
  hist_kernel<<<(NEDGES + 255) / 256, 256, 0, stream>>>(eattr, hist);
  rmax_kernel<<<1, 32, 0, stream>>>(hist, rmax);

  // Encoders
  encoder_kernel<<<(NNODES / 16 + 7) / 8, 256, 0, stream>>>(x, Wp_in, b_in, g_in,
                                                            be_in, h32, states);
  edge_enc_kernel<<<NEDGES / 16 / 8, 256, 0, stream>>>(eattr, rmax, Wp_e, b_e, g_e,
                                                       be_e, e16);
  // Message-passing layers
  for (int l = 0; l < NLAYERS; ++l) {
    zero_kernel<<<((long long)NNODES * HDIM + 255) / 256, 256, 0, stream>>>(
        m, (long long)NNODES * HDIM);
    msg_kernel<<<NEDGES / 16 / 8, 256, 0, stream>>>(
        src, dst, states + (size_t)l * NNODES * HDIM, e16,
        Wp_m1 + (size_t)l * 12 * 8 * 512, msg_b1 + (size_t)l * HDIM,
        Wp_m2 + (size_t)l * 4 * 8 * 512, msg_b2 + (size_t)l * HDIM, m);
    update_kernel<<<(NNODES / 16 + 7) / 8, 256, 0, stream>>>(
        states + (size_t)l * NNODES * HDIM, m, h32,
        Wp_u + (size_t)l * 8 * 8 * 512, upd_b + (size_t)l * HDIM,
        norm_g + (size_t)l * HDIM, norm_b + (size_t)l * HDIM,
        states + (size_t)(l + 1) * NNODES * HDIM);
  }
  // Head + group means
  head_kernel<<<(NNODES / 16 + 7) / 8, 256, 0, stream>>>(states, Wp_jk, b_jk, Wp_o1,
                                                         b_o1, Wp_o2, b_o2, gid, gsum);
  out_kernel<<<(int)(((size_t)NNODES * NCLS + 255) / 256), 256, 0, stream>>>(
      gid, gsum, gcnt, out);
}
